// TSM2_72232759984891
// MI455X (gfx1250) — matrix operand order repaired and checked by reading, not compiled, not run on hardware
//
#include <hip/hip_runtime.h>
#include <hip/hip_bf16.h>
#include <math.h>

typedef __attribute__((ext_vector_type(16))) _Float16 v16h;
typedef __attribute__((ext_vector_type(8)))  float    v8f;

#define EPSc 1e-5f
#define TOKD 4128768   // 16128 * 256

// ---------------- device helpers ----------------
__device__ __forceinline__ float sigmoidf_(float x){ return 1.0f/(1.0f+__expf(-x)); }
__device__ __forceinline__ float softplusf_(float x){ return (x>20.0f)? x : log1pf(__expf(x)); }

// ---------------- WMMA GEMM: C[M,N] = A(f32,row)[M,K] * Bw(f16,row K_pad x N) ----------------
// REQUIREMENTS (guaranteed by host orchestration):
//  * Kdim multiple of 16; every A row readable up to Kpad=(Kdim+31)&~31 columns
//  * Bw rows in [Kdim, Kpad) are zero (so out-of-K A values contribute 0)
// Fused epilogue: +bias[n], +residual[m,n], +pos[(m % pos_period), n], act(1=softplus)
template<int WN>
__global__ __launch_bounds__(128) void gemm_wmma_kernel(
    const float* __restrict__ A, int lda,
    const _Float16* __restrict__ Bw, int ldb,
    const float* __restrict__ bias,
    const float* __restrict__ residual,
    const float* __restrict__ posadd, int pos_period,
    float* __restrict__ C, int ldc,
    int Mrows, int Ncols, int Kdim, int act)
{
  const int wave = threadIdx.x >> 5;
  const int lane = threadIdx.x & 31;
  const int mt_count = Mrows >> 4;
  const int nt_count = Ncols / (16*WN);
  int tile = blockIdx.x * 4 + wave;
  if (tile >= mt_count*nt_count) return;         // whole-wave uniform exit (EXEC stays all-1 for live waves)
  const int mt = tile % mt_count;
  const int nt = tile / mt_count;
  const int m0 = mt << 4;
  const int n0 = nt * (16*WN);
  const int hs = lane >> 4;                      // half-wave select
  const int ml = lane & 15;

  v8f acc[WN];
  #pragma unroll
  for (int i=0;i<WN;++i){
    #pragma unroll
    for (int e=0;e<8;++e) acc[i][e]=0.0f;
  }

  const int Kpad = (Kdim + 31) & ~31;
  // A fragment (16-bit A 16x32 layout): lane ml = M row; half-wave hs selects K runs
  //   a[0..7]  <- k0 + 8*hs      .. +7   (VGPR 0-3, K pairs)
  //   a[8..15] <- k0 + 16 + 8*hs .. +7   (VGPR 4-7, K pairs)
  const float* ap = A + (size_t)(m0+ml)*lda + 8*hs;
  // B fragment: lane holds column n0 + ml; b[0..7] <- rows k0 + 8*hs .. +7, b[8..15] <- rows k0 + 16 + 8*hs .. +7
  const _Float16* bp = Bw + (size_t)(8*hs)*ldb + n0 + ml;
  const size_t bstep = (size_t)32*ldb;

  for (int k0=0;k0<Kpad;k0+=32){
    const float4 f0 = *(const float4*)(ap);
    const float4 f1 = *(const float4*)(ap + 4);
    const float4 f2 = *(const float4*)(ap + 16);
    const float4 f3 = *(const float4*)(ap + 20);
    __builtin_prefetch(ap + 32, 0, 1);           // -> global_prefetch_b8
    __builtin_prefetch(bp + bstep, 0, 1);
    v16h a;
    a[0]=(_Float16)f0.x;  a[1]=(_Float16)f0.y;  a[2]=(_Float16)f0.z;  a[3]=(_Float16)f0.w;
    a[4]=(_Float16)f1.x;  a[5]=(_Float16)f1.y;  a[6]=(_Float16)f1.z;  a[7]=(_Float16)f1.w;
    a[8]=(_Float16)f2.x;  a[9]=(_Float16)f2.y;  a[10]=(_Float16)f2.z; a[11]=(_Float16)f2.w;
    a[12]=(_Float16)f3.x; a[13]=(_Float16)f3.y; a[14]=(_Float16)f3.z; a[15]=(_Float16)f3.w;
    #pragma unroll
    for (int wn=0;wn<WN;++wn){
      v16h b;
      #pragma unroll
      for (int s=0;s<8;++s){ b[s] = bp[16*wn + (size_t)s*ldb]; b[8+s] = bp[16*wn + (size_t)(16+s)*ldb]; }
      acc[wn] = __builtin_amdgcn_wmma_f32_16x16x32_f16(false, a, false, b, (short)0, acc[wn], false, false);
    }
    ap += 32;
    bp += bstep;
  }
  // Epilogue: D layout: VGPR r -> row m0+r (lanes 0-15) / m0+r+8 (lanes 16-31), col n0+16*wn+ml
  #pragma unroll
  for (int wn=0;wn<WN;++wn){
    int col = n0 + 16*wn + ml;
    float bv = bias ? bias[col] : 0.0f;
    #pragma unroll
    for (int r=0;r<8;++r){
      int row = m0 + r + 8*hs;
      float v = acc[wn][r] + bv;
      if (residual) v += residual[(size_t)row*ldc + col];
      if (posadd)   v += posadd[(size_t)(row % pos_period)*ldc + col];
      if (act==1)   v = softplusf_(v);
      C[(size_t)row*ldc + col] = v;
    }
  }
}

// ---------------- weight convert: W (Nout x Kin, f32) -> Wt (Kpad x Nout, f16, zero-padded K) --------
__global__ void convert_wT_kernel(const float* __restrict__ W, _Float16* __restrict__ Wt,
                                  int Nout, int Kin, int Kpad)
{
  int idx = blockIdx.x*blockDim.x + threadIdx.x;
  int total = Kpad*Nout;
  if (idx >= total) return;
  int k = idx / Nout, n = idx % Nout;
  Wt[idx] = (k < Kin) ? (_Float16)W[(size_t)n*Kin + k] : (_Float16)0.0f;
}

// ---------------- series normalization stats (ddof=1 std + 1e-5) ----------------
__global__ void meanstd_kernel(const float* __restrict__ X, float* means, float* stds)
{
  int bm = threadIdx.x;            // 256 threads: bm = b*32+m
  int b = bm >> 5, m = bm & 31;
  const float* p = X + (size_t)b*512*32 + m;
  float s = 0.f;
  for (int l=0;l<512;++l) s += p[(size_t)l*32];
  float mean = s*(1.0f/512.0f);
  float v = 0.f;
  for (int l=0;l<512;++l){ float c = p[(size_t)l*32]-mean; v += c*c; }
  means[bm] = mean;
  stds[bm]  = sqrtf(v*(1.0f/511.0f)) + 1e-5f;
}

// ---------------- patch extraction (normalized) -> A matrix (16128 x 32, cols 16..31 zero) ------
__global__ void patch_kernel(const float* __restrict__ X, const float* __restrict__ means,
                             const float* __restrict__ stds, float* __restrict__ Apat)
{
  int idx = blockIdx.x*blockDim.x + threadIdx.x;
  if (idx >= 16128*32) return;
  int p = idx & 31; int t = idx >> 5;
  float val = 0.0f;
  if (p < 16){
    int n = t % 63;   int bm = t / 63;
    int b = bm >> 5, m = bm & 31;
    val = (X[((size_t)b*512 + n*8 + p)*32 + m] - means[bm]) / stds[bm];
  }
  Apat[idx] = val;
}

// ---------------- tiny softmax ----------------
__global__ void softmax_small_kernel(const float* __restrict__ in, int n, float* __restrict__ out)
{
  if (threadIdx.x != 0 || blockIdx.x != 0) return;
  float mx = -1e30f;
  for (int i=0;i<n;++i) mx = fmaxf(mx, in[i]);
  float e[8]; float s = 0.f;
  for (int i=0;i<n;++i){ e[i] = __expf(in[i]-mx); s += e[i]; }
  for (int i=0;i<n;++i) out[i] = e[i]/s;
}

// ---------------- DSA mixing: dst = sum wA[j]*a[j] + sum wB[j]*c[j] ----------------
struct MixArgs { const float* a[6]; const float* c[6]; };
__global__ void mix_kernel(MixArgs P, const float* __restrict__ wA, int nA,
                           const float* __restrict__ wB, int nB,
                           float* __restrict__ dst, int count)
{
  int i = blockIdx.x*blockDim.x + threadIdx.x;
  if (i >= count) return;
  float s = 0.f;
  for (int j=0;j<nA;++j) s += wA[j]*P.a[j][i];
  for (int j=0;j<nB;++j) s += wB[j]*P.c[j][i];
  dst[i] = s;
}

// ---------------- LayerNorm over D=256 (one block per token, LDS tree reduce) ----------------
__global__ __launch_bounds__(256) void layernorm_kernel(const float* __restrict__ in,
                                                        const float* __restrict__ g,
                                                        const float* __restrict__ b,
                                                        float* __restrict__ out)
{
  __shared__ float red[256];
  int tok = blockIdx.x, d = threadIdx.x;
  float v = in[((size_t)tok<<8) + d];
  red[d] = v; __syncthreads();
  for (int o=128;o>0;o>>=1){ if (d<o) red[d]+=red[d+o]; __syncthreads(); }
  float mean = red[0]*(1.0f/256.0f);
  __syncthreads();
  float c = v - mean;
  red[d] = c*c; __syncthreads();
  for (int o=128;o>0;o>>=1){ if (d<o) red[d]+=red[d+o]; __syncthreads(); }
  float inv = rsqrtf(red[0]*(1.0f/256.0f) + EPSc);
  out[((size_t)tok<<8) + d] = c*inv*g[d] + b[d];
}

// ---------------- LayerNorm over N=63 (per (bm,d)) ----------------
__global__ void layernorm_seq_kernel(const float* __restrict__ in, const float* __restrict__ g,
                                     const float* __restrict__ b, float* __restrict__ out)
{
  int idx = blockIdx.x*blockDim.x + threadIdx.x;
  if (idx >= 256*256) return;
  int bm = idx >> 8, d = idx & 255;
  const float* p = in + (((size_t)bm*63)<<8) + d;
  float s = 0.f;
  for (int n=0;n<63;++n) s += p[(size_t)n<<8];
  float mean = s*(1.0f/63.0f);
  float v = 0.f;
  for (int n=0;n<63;++n){ float c = p[(size_t)n<<8]-mean; v += c*c; }
  float inv = rsqrtf(v*(1.0f/63.0f) + EPSc);
  float* q = out + (((size_t)bm*63)<<8) + d;
  for (int n=0;n<63;++n) q[(size_t)n<<8] = (p[(size_t)n<<8]-mean)*inv*g[n] + b[n];
}

// ---------------- causal depthwise conv (K=4) + SiLU, xh = first half of xz ----------------
__global__ void conv_silu_kernel(const float* __restrict__ xz, const float* __restrict__ cw,
                                 const float* __restrict__ cb, float* __restrict__ xh,
                                 int S, int T)
{
  int idx = blockIdx.x*blockDim.x + threadIdx.x;
  if (idx >= S*512) return;
  int c = idx & 511, s = idx >> 9;
  float w0 = cw[c*4+0], w1 = cw[c*4+1], w2 = cw[c*4+2], w3 = cw[c*4+3], bb = cb[c];
  float x0=0.f, x1=0.f, x2=0.f;
  for (int t=0;t<T;++t){
    size_t tok = (size_t)s*T + t;
    float x3 = xz[tok*1024 + c];
    float acc = x0*w0 + x1*w1 + x2*w2 + x3*w3 + bb;
    xh[tok*512 + c] = acc * sigmoidf_(acc);
    x0=x1; x1=x2; x2=x3;
  }
}

// ---------------- selective scan, fused +xh*D and *silu(z) gating ----------------
__global__ __launch_bounds__(512) void scan_kernel(const float* __restrict__ dt,
                                                   const float* __restrict__ xh,
                                                   const float* __restrict__ xdbl,
                                                   const float* __restrict__ xz,
                                                   const float* __restrict__ A_log,
                                                   const float* __restrict__ Dp,
                                                   float* __restrict__ ybuf,
                                                   int S, int T)
{
  __shared__ float BC[32];          // B (16) then C (16) for current timestep
  int s = blockIdx.x;
  int d = threadIdx.x;              // 0..511
  float Ar[16];
  #pragma unroll
  for (int j=0;j<16;++j) Ar[j] = -__expf(A_log[d*16 + j]);
  float h[16];
  #pragma unroll
  for (int j=0;j<16;++j) h[j] = 0.f;
  float Dv = Dp[d];
  for (int t=0;t<T;++t){
    size_t tok = (size_t)s*T + t;
    if (threadIdx.x < 32) BC[threadIdx.x] = xdbl[tok*48 + 16 + threadIdx.x];
    __syncthreads();
    float dtv = dt[tok*512 + d];
    float xv  = xh[tok*512 + d];
    float zv  = xz[tok*1024 + 512 + d];
    float dx  = dtv * xv;
    float y = 0.f;
    #pragma unroll
    for (int j=0;j<16;++j){
      float dA = __expf(dtv * Ar[j]);
      h[j] = dA*h[j] + dx*BC[j];
      y += h[j]*BC[16+j];
    }
    y += xv*Dv;
    ybuf[tok*512 + d] = y * (zv * sigmoidf_(zv));
    __syncthreads();
  }
}

// ---------------- layout shuffles ----------------
// ch_in (BM,N,D) -> ch_r (B*N, M, D)
__global__ void to_channel_kernel(const float* __restrict__ in, float* __restrict__ out)
{
  int idx = blockIdx.x*blockDim.x + threadIdx.x;
  if (idx >= TOKD) return;
  int d = idx & 255; int q = idx >> 8;
  int m = q % 32; q /= 32;
  int n = q % 63; int b = q / 63;
  out[idx] = in[(((size_t)(b*32+m)*63 + n)<<8) + d];
}
// yC_next[(b*32+m),n,d] = tmp1[(b*63+n),m,d] + tmp2[(b*63+n),31-m,d]
__global__ void from_channel_addflip_kernel(const float* __restrict__ t1,
                                            const float* __restrict__ t2,
                                            float* __restrict__ out)
{
  int idx = blockIdx.x*blockDim.x + threadIdx.x;
  if (idx >= TOKD) return;
  int d = idx & 255; int q = idx >> 8;
  int n = q % 63; int r = q / 63;
  int m = r & 31; int b = r >> 5;
  size_t row = (size_t)(b*63 + n)*32;
  out[idx] = t1[((row + m)<<8) + d] + t2[((row + (31-m))<<8) + d];
}
// time flip: out[s,t,d] = in[s,T-1-t,d]
__global__ void flip_time_kernel(const float* __restrict__ in, float* __restrict__ out, int S, int T)
{
  int idx = blockIdx.x*blockDim.x + threadIdx.x;
  if (idx >= S*T*256) return;
  int d = idx & 255; int q = idx >> 8;
  int t = q % T; int s = q / T;
  out[idx] = in[(((size_t)s*T + (T-1-t))<<8) + d];
}

// ---------------- final de-normalization + (B,M,PRED)->(B,PRED,M) ----------------
__global__ void final_out_kernel(const float* __restrict__ ho, const float* __restrict__ means,
                                 const float* __restrict__ stds, float* __restrict__ out)
{
  int idx = blockIdx.x*blockDim.x + threadIdx.x;
  if (idx >= 8*96*32) return;
  int m = idx & 31; int q = idx >> 5;
  int p = q % 96;  int b = q / 96;
  int bm = b*32 + m;
  out[idx] = ho[(size_t)bm*96 + p]*stds[bm] + means[bm];
}

// ---------------- host orchestration ----------------
struct MW {
  const float *conv_w, *conv_b, *dt_b, *A_log, *Dp;
  _Float16 *in_wT, *x_wT, *dt_wT, *out_wT;
};

static inline void launch_gemm(const float* A,int lda,const _Float16* Bw,int ldb,
  const float* bias,const float* res,const float* posadd,int posT,
  float* C,int ldc,int M,int N,int K,int act,hipStream_t s)
{
  if ((N & 63)==0){
    int tiles = (M>>4)*(N>>6);
    gemm_wmma_kernel<4><<<(tiles+3)/4,128,0,s>>>(A,lda,Bw,ldb,bias,res,posadd,posT,C,ldc,M,N,K,act);
  } else {
    int tiles = (M>>4)*(N>>4);
    gemm_wmma_kernel<1><<<(tiles+3)/4,128,0,s>>>(A,lda,Bw,ldb,bias,res,posadd,posT,C,ldc,M,N,K,act);
  }
}

extern "C" void kernel_launch(void* const* d_in, const int* in_sizes, int n_in,
                              void* d_out, int out_size, void* d_ws, size_t ws_size,
                              hipStream_t stream) {
  (void)in_sizes; (void)n_in; (void)out_size; (void)ws_size;

  // ----- input mapping (setup_inputs dict order, depth-first) -----
  const float* X        = (const float*)d_in[0];
  const float* patch_w  = (const float*)d_in[1];
  const float* patch_b  = (const float*)d_in[2];
  const float* pos      = (const float*)d_in[3];
  auto Lp = [&](int l,int o)->const float* { return (const float*)d_in[4 + 35*l + o]; };
  const float* n1_g = (const float*)d_in[144];
  const float* n1_b = (const float*)d_in[145];
  const float* n2_g = (const float*)d_in[146];
  const float* n2_b = (const float*)d_in[147];
  const float* head_w = (const float*)d_in[148];
  const float* head_b = (const float*)d_in[149];

  // ----- workspace bump allocator -----
  char* base = (char*)d_ws;
  size_t off = 0;
  auto alloc = [&](size_t nbytes)->void* {
    size_t a = (off + 255) & ~(size_t)255; off = a + nbytes; return (void*)(base + a);
  };

  float* means = (float*)alloc(256*4);
  float* stds  = (float*)alloc(256*4);
  float* smax  = (float*)alloc(32*4);
  float* Apat  = (float*)alloc((size_t)16128*32*4);   // lda=32, cols 16..31 zero
  float* yT[5]; float* yC[5];
  yT[0] = (float*)alloc((size_t)TOKD*4); yC[0] = yT[0];
  for (int i=1;i<5;++i){ yT[i]=(float*)alloc((size_t)TOKD*4); yC[i]=(float*)alloc((size_t)TOKD*4); }
  float* mixbuf = (float*)alloc((size_t)TOKD*4);
  float* lnbuf  = (float*)alloc((size_t)TOKD*4);
  float* ch_r   = (float*)alloc((size_t)TOKD*4);
  float* hln    = (float*)alloc((size_t)TOKD*4);
  float* hlnf   = (float*)alloc((size_t)TOKD*4);
  float* tmp1   = (float*)alloc((size_t)TOKD*4);
  float* tmp2   = (float*)alloc((size_t)TOKD*4);
  float* xz     = (float*)alloc((size_t)16128*1024*4);
  float* xh     = (float*)alloc((size_t)16128*512*4);
  float* xdbl   = (float*)alloc((size_t)16128*48*4);
  float* dtb    = (float*)alloc((size_t)16128*512*4);
  float* ybuf   = (float*)alloc((size_t)16128*512*4);
  float* headout= (float*)alloc((size_t)256*96*4);

  auto convW = [&](const float* W, _Float16* Wt, int Nout, int Kin, int Kpad){
    int total = Kpad*Nout;
    convert_wT_kernel<<<(total+255)/256,256,0,stream>>>(W,Wt,Nout,Kin,Kpad);
  };

  // ----- per-mamba f16 transposed weights -----
  MW mw[12];
  const int mbase[3] = {6, 17, 26};   // tok.mamba, ch.fwd, ch.bwd
  for (int l=0;l<4;++l){
    for (int j=0;j<3;++j){
      int b0 = mbase[j];
      MW& w = mw[l*3+j];
      const float* in_w  = Lp(l, b0+0);
      w.conv_w = Lp(l, b0+1); w.conv_b = Lp(l, b0+2);
      const float* x_w   = Lp(l, b0+3);
      const float* dt_w  = Lp(l, b0+4);
      w.dt_b  = Lp(l, b0+5); w.A_log = Lp(l, b0+6); w.Dp = Lp(l, b0+7);
      const float* out_w = Lp(l, b0+8);
      w.in_wT  = (_Float16*)alloc((size_t)256*1024*2);
      w.x_wT   = (_Float16*)alloc((size_t)512*48*2);
      w.dt_wT  = (_Float16*)alloc((size_t)32*512*2);
      w.out_wT = (_Float16*)alloc((size_t)512*256*2);
      convW(in_w,  w.in_wT,  1024, 256, 256);
      convW(x_w,   w.x_wT,   48,   512, 512);
      convW(dt_w,  w.dt_wT,  512,  16,  32);
      convW(out_w, w.out_wT, 256,  512, 512);
    }
  }
  _Float16* patch_wT = (_Float16*)alloc((size_t)32*256*2);
  _Float16* head_wT  = (_Float16*)alloc((size_t)16128*96*2);
  convW(patch_w, patch_wT, 256, 16, 32);
  convW(head_w,  head_wT,  96, 16128, 16128);

  // ----- mamba: S sequences of length T over D=256 -----
  auto run_mamba = [&](const MW& w, const float* xin, int S, int T, float* out, const float* res){
    int MT = S*T;   // always 16128 here
    launch_gemm(xin, 256, w.in_wT, 1024, nullptr, nullptr, nullptr, 0, xz, 1024, MT, 1024, 256, 0, stream);
    { int tot = S*512; conv_silu_kernel<<<(tot+255)/256,256,0,stream>>>(xz, w.conv_w, w.conv_b, xh, S, T); }
    launch_gemm(xh, 512, w.x_wT, 48, nullptr, nullptr, nullptr, 0, xdbl, 48, MT, 48, 512, 0, stream);
    // K=16: A rows (xdbl, lda=48) are readable to col 31; dt_wT rows 16..31 are zero
    launch_gemm(xdbl, 48, w.dt_wT, 512, w.dt_b, nullptr, nullptr, 0, dtb, 512, MT, 512, 16, 1, stream);
    scan_kernel<<<S,512,0,stream>>>(dtb, xh, xdbl, xz, w.A_log, w.Dp, ybuf, S, T);
    launch_gemm(ybuf, 512, w.out_wT, 256, nullptr, res, nullptr, 0, out, 256, MT, 256, 512, 0, stream);
  };

  // ----- stem: normalize, patchify (zero-padded to K=32), embed (+bias +pos fused) -----
  meanstd_kernel<<<1,256,0,stream>>>(X, means, stds);
  patch_kernel<<<(16128*32+255)/256,256,0,stream>>>(X, means, stds, Apat);
  launch_gemm(Apat, 32, patch_wT, 256, patch_b, nullptr, pos, 63, yT[0], 256, 16128, 256, 32, 0, stream);

  const int EW_BLK = (TOKD+255)/256;

  // ----- layers -----
  for (int l=0;l<4;++l){
    softmax_small_kernel<<<1,1,0,stream>>>(Lp(l,0), l+1, smax+0);   // alpha
    softmax_small_kernel<<<1,1,0,stream>>>(Lp(l,1), l+1, smax+8);   // beta
    softmax_small_kernel<<<1,1,0,stream>>>(Lp(l,2), l+2, smax+16);  // theta
    softmax_small_kernel<<<1,1,0,stream>>>(Lp(l,3), l+1, smax+24);  // gamma

    // token mixer
    MixArgs ma = {};
    for (int i=0;i<=l;++i){ ma.a[i]=yT[i]; ma.c[i]=yC[i]; }
    mix_kernel<<<EW_BLK,256,0,stream>>>(ma, smax+0, l+1, smax+8, l+1, mixbuf, TOKD);
    layernorm_kernel<<<16128,256,0,stream>>>(mixbuf, Lp(l,4), Lp(l,5), lnbuf);
    run_mamba(mw[l*3+0], lnbuf, 256, 63, yT[l+1], mixbuf);   // residual fused in out-proj

    // channel mixer
    MixArgs mc = {};
    for (int i=0;i<=l+1;++i) mc.a[i]=yT[i];
    for (int i=0;i<=l;++i)   mc.c[i]=yC[i];
    mix_kernel<<<EW_BLK,256,0,stream>>>(mc, smax+16, l+2, smax+24, l+1, mixbuf, TOKD);
    to_channel_kernel<<<EW_BLK,256,0,stream>>>(mixbuf, ch_r);
    layernorm_kernel<<<16128,256,0,stream>>>(ch_r, Lp(l,15), Lp(l,16), hln);
    run_mamba(mw[l*3+1], hln, 504, 32, tmp1, ch_r);          // fwd, residual = ch_r
    flip_time_kernel<<<EW_BLK,256,0,stream>>>(hln, hlnf, 504, 32);
    run_mamba(mw[l*3+2], hlnf, 504, 32, tmp2, nullptr);      // bwd (flipped domain)
    from_channel_addflip_kernel<<<EW_BLK,256,0,stream>>>(tmp1, tmp2, yC[l+1]);
  }

  // ----- head -----
  layernorm_kernel<<<16128,256,0,stream>>>(yC[4], n1_g, n1_b, lnbuf);
  layernorm_seq_kernel<<<(65536+255)/256,256,0,stream>>>(lnbuf, n2_g, n2_b, mixbuf);
  launch_gemm(mixbuf, 16128, head_wT, 96, head_b, nullptr, nullptr, 0, headout, 96, 256, 96, 16128, 0, stream);
  final_out_kernel<<<(8*96*32+255)/256,256,0,stream>>>(headout, means, stds, (float*)d_out);
}
